// CGCN_71983651881002
// MI455X (gfx1250) — matrix operand order repaired and checked by reading, not compiled, not run on hardware
//
#include <hip/hip_runtime.h>
#include <hip/hip_bf16.h>

typedef __attribute__((ext_vector_type(16))) _Float16 v16h;
typedef __attribute__((ext_vector_type(8)))  float    v8f;

#define UDIM 128
#define NNODES 2848
#define NBATCH 64

// ---------------- utility ----------------
__global__ void zero_kernel(float* __restrict__ p, long n) {
    long i = (long)blockIdx.x * 256 + threadIdx.x;
    if (i < n) p[i] = 0.f;
}

// ---------------- graph normalization ----------------
__global__ void degree_kernel(const int* __restrict__ row, const int* __restrict__ col,
                              const float* __restrict__ w, float* __restrict__ deg, int e) {
    int i = blockIdx.x * 256 + threadIdx.x;
    if (i >= e) return;
    int r = row[i], c = col[i];
    if (r != c) atomicAdd(&deg[r], w[i]);
}

__global__ void norm_kernel(const int* __restrict__ row, const int* __restrict__ col,
                            const float* __restrict__ w, const float* __restrict__ deg,
                            float* __restrict__ nrm, int e) {
    int i = blockIdx.x * 256 + threadIdx.x;
    if (i >= e) return;
    int r = row[i], c = col[i];
    float wr = (r == c) ? 0.f : w[i];
    float dr = deg[r], dc = deg[c];
    float ir = dr > 0.f ? rsqrtf(dr) : 0.f;
    float ic = dc > 0.f ? rsqrtf(dc) : 0.f;
    nrm[i] = -ir * wr * ic;
}

// ---------------- SpMM scatter: dst[col] += scale*norm*src[row] ----------------
// one thread per (edge, 4-channel group); cshift = log2(U/4)
__global__ void spmm_kernel(const float* __restrict__ src, float* __restrict__ dst,
                            const int* __restrict__ row, const int* __restrict__ col,
                            const float* __restrict__ nrm, float scale, int e, int cshift) {
    long idx = (long)blockIdx.x * 256 + threadIdx.x;
    int edge = (int)(idx >> cshift);
    int cv   = (int)(idx & ((1 << cshift) - 1));
    if (edge >= e) return;
    int r = row[edge], c = col[edge];
    float nv = nrm[edge] * scale;
    int u4 = 1 << cshift;
    const float4 s = ((const float4*)src)[(size_t)r * u4 + cv];
    float* d = dst + ((size_t)c * u4 + cv) * 4;
    atomicAdd(d + 0, nv * s.x);
    atomicAdd(d + 1, nv * s.y);
    atomicAdd(d + 2, nv * s.z);
    atomicAdd(d + 3, nv * s.w);
}

// ---------------- layer-0 GEMM (cin=4): scalar VALU, fused BN stats ----------------
__global__ __launch_bounds__(256)
void gemm0_kernel(const float* __restrict__ x, const float* __restrict__ t1,
                  const float* __restrict__ t2, const float* __restrict__ W,
                  const float* __restrict__ bias, float* __restrict__ out,
                  float* __restrict__ stats) {
    __shared__ float red[256];
    int tid = threadIdx.x;
    red[tid] = 0.f;
    __syncthreads();
    long idx = (long)blockIdx.x * 256 + tid;
    int node = (int)(idx >> 7), ch = (int)(idx & 127);
    float v = bias[ch];
#pragma unroll
    for (int k = 0; k < 4; ++k) {
        float a = x[node * 4 + k];
        v += a * W[k * 128 + ch];
        v += t1[node * 4 + k] * W[512 + k * 128 + ch];
        v += (t2[node * 4 + k] - a) * W[1024 + k * 128 + ch];
    }
    out[idx] = v;
    atomicAdd(&red[ch], v);
    atomicAdd(&red[128 + ch], v * v);
    __syncthreads();
    atomicAdd(&stats[tid], red[tid]);
}

// ---------------- weight transpose+convert: W[mat][k][n] f32 -> wt[mat][n][k] f16 ----------------
__global__ void convw_kernel(const float* __restrict__ W, _Float16* __restrict__ wt) {
    int idx = blockIdx.x * 256 + threadIdx.x;   // 3*128*128
    int mat = idx >> 14;
    int k   = (idx >> 7) & 127;
    int n   = idx & 127;
    wt[((size_t)(mat * 128 + n)) * 128 + k] = (_Float16)W[idx];
}

// ---------------- WMMA GEMM: out = A@W0 + T1@W1 + (T2-A)@W2 + b, fused BN stats ----------------
// 256 threads = 8 waves; each wave owns 16 rows x all 128 cols. T2 is read (tx2tmp)
// and overwritten (out) by the same wave's rows only -> in-place safe.
__global__ __launch_bounds__(256)
void gemmU_kernel(const float* __restrict__ A, const float* __restrict__ T1,
                  float* __restrict__ T2, const _Float16* __restrict__ wt,
                  const float* __restrict__ bias, float* __restrict__ stats) {
    __shared__ float red[256];
    int tid = threadIdx.x;
    red[tid] = 0.f;
    __syncthreads();

    int wave = tid >> 5;
    int lane = tid & 31;
    int half = lane >> 4;      // 0 or 1
    int l16  = lane & 15;
    int row0 = blockIdx.x * 128 + wave * 16;
    int m    = row0 + l16;

    v8f acc[8] = {};

    int g     = half * 16;     // A-fragment K offset (lanes 16-31 hold K+16, contiguous)
    int koffB = half * 16;     // B-fragment K offset (lanes 16-31 hold K+16)

    for (int kk = 0; kk < 4; ++kk) {
        int k0 = kk * 32;
        const float* ap  = A  + (size_t)m * UDIM + k0 + g;
        const float* t1p = T1 + (size_t)m * UDIM + k0 + g;
        const float* t2p = T2 + (size_t)m * UDIM + k0 + g;
        v16h fa, f1, f2;
#pragma unroll
        for (int j = 0; j < 8; ++j) {
            float a0 = ap[j],      a1 = ap[8 + j];
            float b0 = t1p[j],     b1 = t1p[8 + j];
            float c0 = t2p[j] - a0, c1 = t2p[8 + j] - a1;
            fa[j] = (_Float16)a0;  fa[8 + j] = (_Float16)a1;
            f1[j] = (_Float16)b0;  f1[8 + j] = (_Float16)b1;
            f2[j] = (_Float16)c0;  f2[8 + j] = (_Float16)c1;
        }
#pragma unroll
        for (int nt = 0; nt < 8; ++nt) {
            int n = nt * 16 + l16;
            const v16h* w0 = (const v16h*)(wt + ((size_t)(0 * 128 + n)) * 128 + k0 + koffB);
            const v16h* w1 = (const v16h*)(wt + ((size_t)(1 * 128 + n)) * 128 + k0 + koffB);
            const v16h* w2 = (const v16h*)(wt + ((size_t)(2 * 128 + n)) * 128 + k0 + koffB);
            acc[nt] = __builtin_amdgcn_wmma_f32_16x16x32_f16(false, fa, false, *w0,
                                                             (short)0, acc[nt], false, false);
            acc[nt] = __builtin_amdgcn_wmma_f32_16x16x32_f16(false, f1, false, *w1,
                                                             (short)0, acc[nt], false, false);
            acc[nt] = __builtin_amdgcn_wmma_f32_16x16x32_f16(false, f2, false, *w2,
                                                             (short)0, acc[nt], false, false);
        }
    }

#pragma unroll
    for (int nt = 0; nt < 8; ++nt) {
        int n = nt * 16 + l16;
        float bn = bias[n];
        float s = 0.f, s2 = 0.f;
#pragma unroll
        for (int r = 0; r < 8; ++r) {
            int mm = row0 + r + half * 8;   // ISA C/D layout: VGPR r -> M=r (+8 for hi lanes)
            float v = acc[nt][r] + bn;
            T2[(size_t)mm * UDIM + n] = v;
            s += v;
            s2 += v * v;
        }
        atomicAdd(&red[n], s);
        atomicAdd(&red[128 + n], s2);
    }
    __syncthreads();
    atomicAdd(&stats[tid], red[tid]);
}

// ---------------- BN finalize + apply ----------------
__global__ void bnfinal_kernel(const float* __restrict__ stats, float* __restrict__ mu,
                               float* __restrict__ rstd, float inv_n) {
    int c = threadIdx.x;  // 128
    float m = stats[c] * inv_n;
    float v = stats[128 + c] * inv_n - m * m;
    mu[c] = m;
    rstd[c] = rsqrtf(fmaxf(v, 0.f) + 1e-5f);
}

__global__ void bnrelu_kernel(const float* __restrict__ out, const float* __restrict__ mu,
                              const float* __restrict__ rstd, const float* __restrict__ gg,
                              const float* __restrict__ bb, float* __restrict__ h, long n4) {
    long i = (long)blockIdx.x * 256 + threadIdx.x;
    if (i >= n4) return;
    int cb = (int)((i & 31) << 2);
    float4 o = ((const float4*)out)[i];
    float4 r;
    r.x = fmaxf((o.x - mu[cb + 0]) * rstd[cb + 0] * gg[cb + 0] + bb[cb + 0], 0.f);
    r.y = fmaxf((o.y - mu[cb + 1]) * rstd[cb + 1] * gg[cb + 1] + bb[cb + 1], 0.f);
    r.z = fmaxf((o.z - mu[cb + 2]) * rstd[cb + 2] * gg[cb + 2] + bb[cb + 2], 0.f);
    r.w = fmaxf((o.w - mu[cb + 3]) * rstd[cb + 3] * gg[cb + 3] + bb[cb + 3], 0.f);
    ((float4*)h)[i] = r;
}

// ---------------- readout heads ----------------
__global__ void node_logits_kernel(const float* __restrict__ h, const float* __restrict__ nw,
                                   const float* __restrict__ nb, float* __restrict__ out) {
    int tid = threadIdx.x;
    int node = blockIdx.x * 8 + (tid >> 5);
    int lane = tid & 31;
    float4 a = ((const float4*)h)[(size_t)node * 32 + lane];
    float4 w = ((const float4*)nw)[lane];
    float acc = a.x * w.x + a.y * w.y + a.z * w.z + a.w * w.w;
    for (int o = 16; o > 0; o >>= 1) acc += __shfl_down(acc, o, 32);
    if (lane == 0) out[node] = acc + nb[0];
}

__global__ void graph_logits_kernel(const float* __restrict__ h, const float* __restrict__ gw,
                                    const float* __restrict__ gb, float* __restrict__ out) {
    __shared__ float red[256];
    int b = blockIdx.x, tid = threadIdx.x;
    const float4* hp = (const float4*)(h + (size_t)b * (NNODES * UDIM));
    const float4* wp = (const float4*)gw;
    const int n4 = NNODES * UDIM / 4;
    float acc = 0.f;
    for (int i = tid; i < n4; i += 256) {
        float4 a = hp[i], w = wp[i];
        acc += a.x * w.x + a.y * w.y + a.z * w.z + a.w * w.w;
    }
    red[tid] = acc;
    __syncthreads();
    for (int s = 128; s > 0; s >>= 1) {
        if (tid < s) red[tid] += red[tid + s];
        __syncthreads();
    }
    if (tid == 0) out[b] = red[0] + gb[0];
}

extern "C" void kernel_launch(void* const* d_in, const int* in_sizes, int n_in,
                              void* d_out, int out_size, void* d_ws, size_t ws_size,
                              hipStream_t stream) {
    const float* x  = (const float*)d_in[0];
    const int*   ei = (const int*)d_in[1];
    const float* w  = (const float*)d_in[2];
    const int E    = in_sizes[2];
    const int NTOT = in_sizes[3];            // 182272 (multiple of 128)
    const int* row = ei;
    const int* col = ei + E;

    float* ws = (float*)d_ws;
    size_t NU = (size_t)NTOT * UDIM;
    float* Ab    = ws;                // activations (tx0 / h)
    float* T1    = Ab + NU;           // tx1
    float* T2    = T1 + NU;           // tx2tmp, then GEMM output (in-place)
    float* deg   = T2 + NU;
    float* nrm   = deg + NTOT;
    float* stats = nrm + E;           // [sum(128), sumsq(128)]
    float* mu    = stats + 256;
    float* rstd  = mu + 128;
    _Float16* wt = (_Float16*)(rstd + 128);  // 3*128*128 f16, transposed [mat][n][k]

#define ZERO(p, n) zero_kernel<<<(int)(((long)(n) + 255) / 256), 256, 0, stream>>>((p), (long)(n))

    // normalization: deg = segsum(w by row, no self loops); norm = -d^-1/2 w d^-1/2
    ZERO(deg, NTOT);
    degree_kernel<<<(E + 255) / 256, 256, 0, stream>>>(row, col, w, deg, E);
    norm_kernel<<<(E + 255) / 256, 256, 0, stream>>>(row, col, w, deg, nrm, E);

    // ---- layer 0 (cin = 4) ----
    float* t1s = T1;
    float* t2s = T1 + (size_t)NTOT * 4;
    ZERO(T1, (long)NTOT * 8);
    spmm_kernel<<<(E + 255) / 256, 256, 0, stream>>>(x, t1s, row, col, nrm, 1.0f, E, 0);
    spmm_kernel<<<(E + 255) / 256, 256, 0, stream>>>(t1s, t2s, row, col, nrm, 2.0f, E, 0);
    ZERO(stats, 256);
    gemm0_kernel<<<NTOT * UDIM / 256, 256, 0, stream>>>(
        x, t1s, t2s, (const float*)d_in[4], (const float*)d_in[5], T2, stats);
    bnfinal_kernel<<<1, 128, 0, stream>>>(stats, mu, rstd, 1.0f / NTOT);
    bnrelu_kernel<<<NTOT * 32 / 256, 256, 0, stream>>>(
        T2, mu, rstd, (const float*)d_in[6], (const float*)d_in[7], Ab, (long)NTOT * 32);

    // ---- layers 1..4 (U = 128), WMMA path ----
    long spmm_threads = (long)E * 32;
    int  spmm_blocks  = (int)((spmm_threads + 255) / 256);
    for (int i = 1; i < 5; ++i) {
        const float* Wi  = (const float*)d_in[4 + 4 * i];
        const float* bi  = (const float*)d_in[5 + 4 * i];
        const float* gi  = (const float*)d_in[6 + 4 * i];
        const float* bti = (const float*)d_in[7 + 4 * i];
        convw_kernel<<<192, 256, 0, stream>>>(Wi, wt);
        ZERO(T1, NU);
        spmm_kernel<<<spmm_blocks, 256, 0, stream>>>(Ab, T1, row, col, nrm, 1.0f, E, 5);
        ZERO(T2, NU);
        spmm_kernel<<<spmm_blocks, 256, 0, stream>>>(T1, T2, row, col, nrm, 2.0f, E, 5);
        ZERO(stats, 256);
        gemmU_kernel<<<NTOT / 128, 256, 0, stream>>>(Ab, T1, T2, wt, bi, stats);
        bnfinal_kernel<<<1, 128, 0, stream>>>(stats, mu, rstd, 1.0f / NTOT);
        bnrelu_kernel<<<NTOT * 32 / 256, 256, 0, stream>>>(T2, mu, rstd, gi, bti, Ab,
                                                           (long)NTOT * 32);
    }

    // ---- heads ----
    float* out = (float*)d_out;
    node_logits_kernel<<<NTOT / 8, 256, 0, stream>>>(
        Ab, (const float*)d_in[24], (const float*)d_in[25], out);
    graph_logits_kernel<<<NBATCH, 256, 0, stream>>>(
        Ab, (const float*)d_in[26], (const float*)d_in[27], out + NTOT);
#undef ZERO
}
